// Generator_53102975648085
// MI455X (gfx1250) — matrix operand order repaired and checked by reading, not compiled, not run on hardware
//
#include <hip/hip_runtime.h>

// ---------------- model constants ----------------
#define BB 16
#define SS 192
#define DD 512
#define NHH 2
#define HD 256
#define NLL 4
#define FFE 4
#define DDEC 3
#define NTOK (BB*SS)          // 3072

typedef _Float16 half_t;
typedef __attribute__((ext_vector_type(16))) _Float16 v16h;
typedef __attribute__((ext_vector_type(8)))  float    v8f;

// ---------------- wave reductions (wave32) ----------------
__device__ inline float wred_sum(float v){
#pragma unroll
  for (int o = 16; o > 0; o >>= 1) v += __shfl_down(v, o, 32);
  return v;
}
__device__ inline float wred_max(float v){
#pragma unroll
  for (int o = 16; o > 0; o >>= 1) v = fmaxf(v, __shfl_down(v, o, 32));
  return v;
}
// block sum for 256-thread blocks (8 waves); reentrant
__device__ inline float blk_sum256(float v, float* wp, float* bc){
  int t = threadIdx.x, lane = t & 31, wid = t >> 5;
  float s = wred_sum(v);
  if (lane == 0) wp[wid] = s;
  __syncthreads();
  if (t == 0){ float a = 0.f; for (int i = 0; i < 8; i++) a += wp[i]; *bc = a; }
  __syncthreads();
  float r = *bc;
  __syncthreads();
  return r;
}

// ---------------- WMMA fragment loaders (contiguous k order, same in A and B) ----------------
// A 16x32 f16: lane L: M = m0 + (L&15), hi = L>>4.
//   e in [0,16): K = k0 + hi*16 + e
__device__ inline v16h load_a_frag(const half_t* A, int lda, int m0, int k0){
  int lane = threadIdx.x & 31;
  int m = m0 + (lane & 15);
  int hi = lane >> 4;
  const half_t* p = A + (size_t)m * lda + k0 + hi * 16;
  v16h f;
#pragma unroll
  for (int e = 0; e < 16; e++) f[e] = p[e];
  return f;
}
// B 32x16 f16, B[k,n] = W[n*ldw + k] (k-contiguous per column):
//   lane L: N = n0 + (L&15), K = k0 + (L>>4)*16 + e
__device__ inline v16h load_b_kmaj(const half_t* W, int ldw, int n0, int k0){
  int lane = threadIdx.x & 31;
  int n = n0 + (lane & 15);
  int hi = lane >> 4;
  const half_t* p = W + (size_t)n * ldw + k0 + hi * 16;
  v16h f;
#pragma unroll
  for (int e = 0; e < 16; e++) f[e] = p[e];
  return f;
}
// B 32x16 f16, B[k,n] = V[k*ldv + n] (row-major K x N source)
__device__ inline v16h load_b_nmaj(const half_t* V, int ldv, int n0, int k0){
  int lane = threadIdx.x & 31;
  int n = n0 + (lane & 15);
  int hi = lane >> 4;
  v16h f;
#pragma unroll
  for (int e = 0; e < 16; e++) f[e] = V[(size_t)(k0 + hi * 16 + e) * ldv + n];
  return f;
}
// C/D 16x16 f32 store: VGPR r holds M = r + 8*(lane>>4), N = lane&15
__device__ inline void store_tile(float* Cf, half_t* Ch, int ldc, int m0, int n0,
                                  const float* bias, v8f acc){
  int lane = threadIdx.x & 31;
  int n = n0 + (lane & 15);
  int hi = lane >> 4;
  float bv = bias ? bias[n] : 0.0f;
#pragma unroll
  for (int r = 0; r < 8; r++){
    int m = m0 + r + hi * 8;
    float v = acc[r] + bv;
    if (Cf) Cf[(size_t)m * ldc + n] = v;
    if (Ch) Ch[(size_t)m * ldc + n] = (half_t)v;
  }
}

#define WMMA_F16(a, b, c) \
  __builtin_amdgcn_wmma_f32_16x16x32_f16(false, (a), false, (b), (short)0, (c), false, false)

// ---------------- kernels ----------------
__global__ void convert_w_kernel(const float* qkv, const float* ow,
                                 half_t* qkvh, half_t* owh){
  const int n1 = NLL * 3 * DD * DD;      // 4*1536*512
  const int n2 = NLL * DD * DD;          // 4*512*512
  for (int i = blockIdx.x * blockDim.x + threadIdx.x; i < n1 + n2;
       i += gridDim.x * blockDim.x){
    if (i < n1) qkvh[i] = (half_t)qkv[i];
    else        owh[i - n1] = (half_t)ow[i - n1];
  }
}

// embed + batch-indexed positional encoding (faithful to reference bug)
__global__ __launch_bounds__(256)
void embed_kernel(const float* src, const float* wemb, const float* semb,
                  float* X, half_t* Xh){
  int n = blockIdx.x;                 // token (b*S + s)
  int bidx = n / SS;
  const float* row = src + (size_t)n * 258;
  int wi = (int)row[256];
  int si = (int)row[257];
  for (int d = threadIdx.x; d < DD; d += 256){
    float v;
    if (d < 256)      v = row[d];
    else if (d < 384) v = wemb[(size_t)wi * 128 + (d - 256)];
    else              v = semb[(size_t)si * 128 + (d - 384)];
    float div = __expf((float)(d & ~1) * (-9.210340371976184f / 512.0f));
    float ang = (float)bidx * div;
    v += (d & 1) ? cosf(ang) : sinf(ang);
    X[(size_t)n * DD + d] = v;
    Xh[(size_t)n * DD + d] = (half_t)v;
  }
}

// generic GEMM: C[m,n] = sum_k A[m,k]*W[n,k] + bias[n]
// 4 waves/block, each wave computes a 32x32 output (2x2 WMMA tiles) -> block 64x64
__global__ __launch_bounds__(128)
void gemm_kernel(const half_t* A, int lda, const half_t* W, int ldw,
                 const float* bias, float* Cf, half_t* Ch, int ldc, int K){
  int w = threadIdx.x >> 5;
  int m0 = blockIdx.x * 64 + (w >> 1) * 32;
  int n0 = blockIdx.y * 64 + (w & 1) * 32;
  v8f a00 = {}, a01 = {}, a10 = {}, a11 = {};
  for (int k0 = 0; k0 < K; k0 += 32){
    v16h fa0 = load_a_frag(A, lda, m0, k0);
    v16h fa1 = load_a_frag(A, lda, m0 + 16, k0);
    v16h fb0 = load_b_kmaj(W, ldw, n0, k0);
    v16h fb1 = load_b_kmaj(W, ldw, n0 + 16, k0);
    a00 = WMMA_F16(fa0, fb0, a00);
    a01 = WMMA_F16(fa0, fb1, a01);
    a10 = WMMA_F16(fa1, fb0, a10);
    a11 = WMMA_F16(fa1, fb1, a11);
  }
  store_tile(Cf, Ch, ldc, m0,      n0,      bias, a00);
  store_tile(Cf, Ch, ldc, m0,      n0 + 16, bias, a01);
  store_tile(Cf, Ch, ldc, m0 + 16, n0,      bias, a10);
  store_tile(Cf, Ch, ldc, m0 + 16, n0 + 16, bias, a11);
}

// per-(b,h) scores = Q @ K^T (raw; 1/16 scale applied in softmax)
__global__ __launch_bounds__(128)
void score_kernel(const half_t* QKV, float* SC){
  int c = blockIdx.z, b = c >> 1, h = c & 1;
  const half_t* Qb = QKV + (size_t)(b * SS) * (3 * DD) + h * HD;
  const half_t* Kb = Qb + DD;
  int w = threadIdx.x >> 5;
  int m0 = blockIdx.x * 64 + (w >> 1) * 32;
  int n0 = blockIdx.y * 64 + (w & 1) * 32;
  v8f a00 = {}, a01 = {}, a10 = {}, a11 = {};
  for (int k0 = 0; k0 < HD; k0 += 32){
    v16h fa0 = load_a_frag(Qb, 3 * DD, m0, k0);
    v16h fa1 = load_a_frag(Qb, 3 * DD, m0 + 16, k0);
    v16h fb0 = load_b_kmaj(Kb, 3 * DD, n0, k0);
    v16h fb1 = load_b_kmaj(Kb, 3 * DD, n0 + 16, k0);
    a00 = WMMA_F16(fa0, fb0, a00);
    a01 = WMMA_F16(fa0, fb1, a01);
    a10 = WMMA_F16(fa1, fb0, a10);
    a11 = WMMA_F16(fa1, fb1, a11);
  }
  float* Cb = SC + (size_t)c * SS * SS;
  store_tile(Cb, nullptr, SS, m0,      n0,      nullptr, a00);
  store_tile(Cb, nullptr, SS, m0,      n0 + 16, nullptr, a01);
  store_tile(Cb, nullptr, SS, m0 + 16, n0,      nullptr, a10);
  store_tile(Cb, nullptr, SS, m0 + 16, n0 + 16, nullptr, a11);
}

__global__ __launch_bounds__(192)
void softmax_kernel(const float* SC, half_t* Ph){
  int r = blockIdx.x;               // 0..6143  (c*192 + qi)
  int t = threadIdx.x, lane = t & 31, wid = t >> 5;
  __shared__ float wp[6];
  __shared__ float bc;
  float s = SC[(size_t)r * SS + t] * 0.0625f;   // / sqrt(256)
  float m = wred_max(s);
  if (lane == 0) wp[wid] = m;
  __syncthreads();
  if (t == 0){ float mm = wp[0]; for (int i = 1; i < 6; i++) mm = fmaxf(mm, wp[i]); bc = mm; }
  __syncthreads();
  float e = __expf(s - bc);
  float ss = wred_sum(e);
  if (lane == 0) wp[wid] = ss;
  __syncthreads();
  if (t == 0){ float a = 0.f; for (int i = 0; i < 6; i++) a += wp[i]; bc = a; }
  __syncthreads();
  Ph[(size_t)r * SS + t] = (half_t)(e / bc);
}

// ATT[b, qi, h*256+dv] = P @ V
__global__ __launch_bounds__(128)
void pv_kernel(const half_t* P, const half_t* QKV, half_t* ATT){
  int c = blockIdx.z, b = c >> 1, h = c & 1;
  const half_t* Pb = P + (size_t)c * SS * SS;
  const half_t* Vb = QKV + (size_t)(b * SS) * (3 * DD) + 2 * DD + h * HD;
  half_t* Ob = ATT + (size_t)(b * SS) * DD + h * HD;
  int w = threadIdx.x >> 5;
  int m0 = blockIdx.x * 64 + (w >> 1) * 32;
  int n0 = blockIdx.y * 64 + (w & 1) * 32;
  v8f a00 = {}, a01 = {}, a10 = {}, a11 = {};
  for (int k0 = 0; k0 < SS; k0 += 32){
    v16h fa0 = load_a_frag(Pb, SS, m0, k0);
    v16h fa1 = load_a_frag(Pb, SS, m0 + 16, k0);
    v16h fb0 = load_b_nmaj(Vb, 3 * DD, n0, k0);
    v16h fb1 = load_b_nmaj(Vb, 3 * DD, n0 + 16, k0);
    a00 = WMMA_F16(fa0, fb0, a00);
    a01 = WMMA_F16(fa0, fb1, a01);
    a10 = WMMA_F16(fa1, fb0, a10);
    a11 = WMMA_F16(fa1, fb1, a11);
  }
  store_tile(nullptr, Ob, DD, m0,      n0,      nullptr, a00);
  store_tile(nullptr, Ob, DD, m0,      n0 + 16, nullptr, a01);
  store_tile(nullptr, Ob, DD, m0 + 16, n0,      nullptr, a10);
  store_tile(nullptr, Ob, DD, m0 + 16, n0 + 16, nullptr, a11);
}

// X = LN(X + R) ; also emit f16 copy
__global__ __launch_bounds__(256)
void add_ln_kernel(float* X, const float* R, const float* g, const float* bb,
                   half_t* Xh){
  __shared__ float wp[8]; __shared__ float bc;
  int n = blockIdx.x, t = threadIdx.x;
  size_t base = (size_t)n * DD;
  float v0 = X[base + t] + R[base + t];
  float v1 = X[base + 256 + t] + R[base + 256 + t];
  float mean = blk_sum256(v0 + v1, wp, &bc) * (1.0f / 512.0f);
  float d0 = v0 - mean, d1 = v1 - mean;
  float var = blk_sum256(d0 * d0 + d1 * d1, wp, &bc) * (1.0f / 512.0f);
  float r = rsqrtf(var + 1e-5f);
  float o0 = d0 * r * g[t] + bb[t];
  float o1 = d1 * r * g[256 + t] + bb[256 + t];
  X[base + t] = o0; X[base + 256 + t] = o1;
  Xh[base + t] = (half_t)o0; Xh[base + 256 + t] = (half_t)o1;
}

// X = LN2(X + relu(X@w1^T+b1)@w2^T+b2)   (FF = 4, done in VALU)
__global__ __launch_bounds__(256)
void ffn_ln_kernel(float* X, const float* w1, const float* b1,
                   const float* w2, const float* b2,
                   const float* g, const float* bb, half_t* Xh){
  __shared__ float wp[8]; __shared__ float bc; __shared__ float h[FFE];
  int n = blockIdx.x, t = threadIdx.x;
  size_t base = (size_t)n * DD;
  float x0 = X[base + t], x1 = X[base + 256 + t];
#pragma unroll
  for (int j = 0; j < FFE; j++){
    float p = x0 * w1[j * DD + t] + x1 * w1[j * DD + 256 + t];
    float s = blk_sum256(p, wp, &bc);
    if (t == 0) h[j] = fmaxf(s + b1[j], 0.0f);
  }
  __syncthreads();
  float f0 = b2[t], f1 = b2[256 + t];
#pragma unroll
  for (int j = 0; j < FFE; j++){
    f0 += h[j] * w2[t * FFE + j];
    f1 += h[j] * w2[(256 + t) * FFE + j];
  }
  float v0 = x0 + f0, v1 = x1 + f1;
  float mean = blk_sum256(v0 + v1, wp, &bc) * (1.0f / 512.0f);
  float d0 = v0 - mean, d1 = v1 - mean;
  float var = blk_sum256(d0 * d0 + d1 * d1, wp, &bc) * (1.0f / 512.0f);
  float r = rsqrtf(var + 1e-5f);
  float o0 = d0 * r * g[t] + bb[t];
  float o1 = d1 * r * g[256 + t] + bb[256 + t];
  X[base + t] = o0; X[base + 256 + t] = o1;
  Xh[base + t] = (half_t)o0; Xh[base + 256 + t] = (half_t)o1;
}

// final LN + 3-dim projection -> MEM ; also fill out cols 3,4 and zero row s==0
__global__ __launch_bounds__(256)
void final_mem_kernel(const float* X, const float* g, const float* bb,
                      const float* ew, const float* eb, const float* src,
                      float* MEMO, float* out){
  __shared__ float wp[8]; __shared__ float bc;
  int n = blockIdx.x, t = threadIdx.x;
  size_t base = (size_t)n * DD;
  float v0 = X[base + t], v1 = X[base + 256 + t];
  float mean = blk_sum256(v0 + v1, wp, &bc) * (1.0f / 512.0f);
  float d0 = v0 - mean, d1 = v1 - mean;
  float var = blk_sum256(d0 * d0 + d1 * d1, wp, &bc) * (1.0f / 512.0f);
  float r = rsqrtf(var + 1e-5f);
  float y0 = d0 * r * g[t] + bb[t];
  float y1 = d1 * r * g[256 + t] + bb[256 + t];
#pragma unroll
  for (int j = 0; j < DDEC; j++){
    float s = blk_sum256(y0 * ew[j * DD + t] + y1 * ew[j * DD + 256 + t], wp, &bc);
    if (t == 0) MEMO[(size_t)n * DDEC + j] = s + eb[j];
  }
  if (t == 0){
    out[(size_t)n * 5 + 3] = src[(size_t)n * 258 + 256];
    out[(size_t)n * 5 + 4] = src[(size_t)n * 258 + 257];
    if (n % SS == 0){
      out[(size_t)n * 5 + 0] = 0.f; out[(size_t)n * 5 + 1] = 0.f;
      out[(size_t)n * 5 + 2] = 0.f;
    }
  }
}

// --------- incremental autoregressive decoder: one block per batch element ---------
__device__ inline void ln3v(const float* a, const float* g, const float* b, float* o){
  float m = (a[0] + a[1] + a[2]) * (1.0f / 3.0f);
  float d0 = a[0] - m, d1 = a[1] - m, d2 = a[2] - m;
  float var = (d0 * d0 + d1 * d1 + d2 * d2) * (1.0f / 3.0f);
  float r = rsqrtf(var + 1e-5f);
  o[0] = d0 * r * g[0] + b[0];
  o[1] = d1 * r * g[1] + b[1];
  o[2] = d2 * r * g[2] + b[2];
}

// softmax + weighted-V over up to 192 keys; result (3) in oout (shared)
__device__ inline void attn_core(float sc, bool act, const float* v3,
                                 float (*wp)[4], float* bcmax, float* oout){
  int t = threadIdx.x, lane = t & 31, wid = t >> 5;
  float m = wred_max(sc);
  if (lane == 0) wp[wid][0] = m;
  __syncthreads();
  if (t == 0){
    float mm = wp[0][0];
    for (int i = 1; i < 6; i++) mm = fmaxf(mm, wp[i][0]);
    *bcmax = mm;
  }
  __syncthreads();
  float e = act ? __expf(sc - *bcmax) : 0.0f;
  float p1 = 0.f, p2 = 0.f, p3 = 0.f;
  if (act){ p1 = e * v3[0]; p2 = e * v3[1]; p3 = e * v3[2]; }
  float s0 = wred_sum(e), s1 = wred_sum(p1), s2 = wred_sum(p2), s3 = wred_sum(p3);
  if (lane == 0){ wp[wid][0] = s0; wp[wid][1] = s1; wp[wid][2] = s2; wp[wid][3] = s3; }
  __syncthreads();
  if (t == 0){
    float a0 = 0, a1 = 0, a2 = 0, a3 = 0;
    for (int i = 0; i < 6; i++){
      a0 += wp[i][0]; a1 += wp[i][1]; a2 += wp[i][2]; a3 += wp[i][3];
    }
    float inv = 1.0f / a0;
    oout[0] = a1 * inv; oout[1] = a2 * inv; oout[2] = a3 * inv;
  }
  __syncthreads();
}

__global__ __launch_bounds__(192)
void decoder_kernel(const float* MEMO,
                    const float* saqkvw, const float* saqkvb,
                    const float* saow, const float* saob,
                    const float* caqkvw, const float* caqkvb,
                    const float* caow, const float* caob,
                    const float* ln1g, const float* ln1b,
                    const float* ln2g, const float* ln2b,
                    const float* ln3g, const float* ln3b,
                    const float* ff1w, const float* ff1b,
                    const float* ff2w, const float* ff2b,
                    float* out){
  int b = blockIdx.x, t = threadIdx.x;
  __shared__ float Kc[NLL][SS][3], Vc[NLL][SS][3];    // cross-attn K/V (precomputed)
  __shared__ float kca[NLL][SS][3], vca[NLL][SS][3];  // self-attn K/V caches
  __shared__ float DW[580];
  __shared__ float yv[3], qv[3], oo[3], hsh[FFE];
  __shared__ float wp[6][4];
  __shared__ float bcm;

  // weight segment offsets in DW
  const int offs[18] = {0,108,144,180,192,300,336,372,384,396,408,420,432,444,456,504,520,568};
  const int cnts[18] = {108,36,36,12,108,36,36,12,12,12,12,12,12,12,48,16,48,12};
  const float* srcs[18] = {saqkvw,saqkvb,saow,saob,caqkvw,caqkvb,caow,caob,
                           ln1g,ln1b,ln2g,ln2b,ln3g,ln3b,ff1w,ff1b,ff2w,ff2b};
  for (int seg = 0; seg < 18; seg++)
    for (int i = t; i < cnts[seg]; i += 192) DW[offs[seg] + i] = srcs[seg][i];
  __syncthreads();

  // precompute cross-attn K/V from memory (fixed per step)
  const float* mb = MEMO + (size_t)b * SS * DDEC;
  for (int idx = t; idx < NLL * SS; idx += 192){
    int l = idx / SS, p = idx % SS;
    const float* w = DW + offs[4] + l * 27;
    const float* bi = DW + offs[5] + l * 9;
#pragma unroll
    for (int j = 0; j < 3; j++){
      float kv = bi[3 + j], vv = bi[6 + j];
#pragma unroll
      for (int d = 0; d < 3; d++){
        kv += mb[p * 3 + d] * w[(3 + j) * 3 + d];
        vv += mb[p * 3 + d] * w[(6 + j) * 3 + d];
      }
      Kc[l][p][j] = kv; Vc[l][p][j] = vv;
    }
  }
  if (t < 3) yv[t] = 0.0f;   // buf row 0 is zeros
  __syncthreads();

  const float isq3 = 0.57735026918962576f;  // 1/sqrt(3)

  for (int i = 0; i < SS - 1; i++){
    for (int l = 0; l < NLL; l++){
      // ---- masked self-attention (incremental; keys 0..i) ----
      if (t < 9){
        const float* w = DW + offs[0] + l * 27;
        const float* bi = DW + offs[1] + l * 9;
        float v = bi[t];
#pragma unroll
        for (int d = 0; d < 3; d++) v += yv[d] * w[t * 3 + d];
        if (t < 3)      qv[t] = v;
        else if (t < 6) kca[l][i][t - 3] = v;
        else            vca[l][i][t - 6] = v;
      }
      __syncthreads();
      float sc = -3.0e38f;
      bool act = (t <= i);
      if (act)
        sc = (qv[0]*kca[l][t][0] + qv[1]*kca[l][t][1] + qv[2]*kca[l][t][2]) * isq3;
      attn_core(sc, act, &vca[l][t][0], wp, &bcm, oo);
      if (t == 0){
        const float* w = DW + offs[2] + l * 9;
        const float* bi = DW + offs[3] + l * 3;
        float a[3];
#pragma unroll
        for (int j = 0; j < 3; j++)
          a[j] = yv[j] + bi[j] + oo[0]*w[j*3+0] + oo[1]*w[j*3+1] + oo[2]*w[j*3+2];
        ln3v(a, DW + offs[8] + l * 3, DW + offs[9] + l * 3, yv);
      }
      __syncthreads();
      // ---- cross-attention (keys = 192 memory positions) ----
      if (t < 3){
        const float* w = DW + offs[4] + l * 27;
        const float* bi = DW + offs[5] + l * 9;
        qv[t] = bi[t] + yv[0]*w[t*3+0] + yv[1]*w[t*3+1] + yv[2]*w[t*3+2];
      }
      __syncthreads();
      float sc2 = (qv[0]*Kc[l][t][0] + qv[1]*Kc[l][t][1] + qv[2]*Kc[l][t][2]) * isq3;
      attn_core(sc2, true, &Vc[l][t][0], wp, &bcm, oo);
      if (t == 0){
        const float* w = DW + offs[6] + l * 9;
        const float* bi = DW + offs[7] + l * 3;
        float a[3];
#pragma unroll
        for (int j = 0; j < 3; j++)
          a[j] = yv[j] + bi[j] + oo[0]*w[j*3+0] + oo[1]*w[j*3+1] + oo[2]*w[j*3+2];
        ln3v(a, DW + offs[10] + l * 3, DW + offs[11] + l * 3, yv);
      }
      __syncthreads();
      // ---- FFN (3 -> 4 -> 3) ----
      if (t < FFE){
        const float* w = DW + offs[14] + l * 12;
        const float* bi = DW + offs[15] + l * 4;
        hsh[t] = fmaxf(bi[t] + yv[0]*w[t*3+0] + yv[1]*w[t*3+1] + yv[2]*w[t*3+2], 0.0f);
      }
      __syncthreads();
      if (t == 0){
        const float* w = DW + offs[16] + l * 12;
        const float* bi = DW + offs[17] + l * 3;
        float a[3];
#pragma unroll
        for (int j = 0; j < 3; j++)
          a[j] = yv[j] + bi[j] + hsh[0]*w[j*4+0] + hsh[1]*w[j*4+1]
               + hsh[2]*w[j*4+2] + hsh[3]*w[j*4+3];
        ln3v(a, DW + offs[12] + l * 3, DW + offs[13] + l * 3, yv);
      }
      __syncthreads();
    }
    // buf[i+1] = decoder_out[i] ; yv carries into next step as y0
    if (t < 3) out[(size_t)(b * SS + i + 1) * 5 + t] = yv[t];
    __syncthreads();
  }
}

// ---------------- host-side orchestration ----------------
extern "C" void kernel_launch(void* const* d_in, const int* in_sizes, int n_in,
                              void* d_out, int out_size, void* d_ws, size_t ws_size,
                              hipStream_t stream){
  (void)in_sizes; (void)n_in; (void)out_size; (void)ws_size;
  const float* src      = (const float*)d_in[0];
  const float* word_emb = (const float*)d_in[1];
  const float* syll_emb = (const float*)d_in[2];
  const float* eqkvw = (const float*)d_in[3];
  const float* eqkvb = (const float*)d_in[4];
  const float* eow   = (const float*)d_in[5];
  const float* eob   = (const float*)d_in[6];
  const float* eln1g = (const float*)d_in[7];
  const float* eln1b = (const float*)d_in[8];
  const float* eff1w = (const float*)d_in[9];
  const float* eff1b = (const float*)d_in[10];
  const float* eff2w = (const float*)d_in[11];
  const float* eff2b = (const float*)d_in[12];
  const float* eln2g = (const float*)d_in[13];
  const float* eln2b = (const float*)d_in[14];
  const float* normfg = (const float*)d_in[15];
  const float* normfb = (const float*)d_in[16];
  const float* eoutw = (const float*)d_in[17];
  const float* eoutb = (const float*)d_in[18];
  const float* dsaqkvw = (const float*)d_in[19];
  const float* dsaqkvb = (const float*)d_in[20];
  const float* dsaow = (const float*)d_in[21];
  const float* dsaob = (const float*)d_in[22];
  const float* dcaqkvw = (const float*)d_in[23];
  const float* dcaqkvb = (const float*)d_in[24];
  const float* dcaow = (const float*)d_in[25];
  const float* dcaob = (const float*)d_in[26];
  const float* dln1g = (const float*)d_in[27];
  const float* dln1b = (const float*)d_in[28];
  const float* dln2g = (const float*)d_in[29];
  const float* dln2b = (const float*)d_in[30];
  const float* dln3g = (const float*)d_in[31];
  const float* dln3b = (const float*)d_in[32];
  const float* dff1w = (const float*)d_in[33];
  const float* dff1b = (const float*)d_in[34];
  const float* dff2w = (const float*)d_in[35];
  const float* dff2b = (const float*)d_in[36];
  float* out = (float*)d_out;

  char* ws = (char*)d_ws;
  size_t off = 0;
  auto alloc = [&](size_t bytes) -> void* {
    off = (off + 255) & ~(size_t)255;
    void* p = ws + off;
    off += bytes;
    return p;
  };
  float*  X     = (float*) alloc((size_t)NTOK * DD * 4);
  half_t* Xh    = (half_t*)alloc((size_t)NTOK * DD * 2);
  half_t* QKVh  = (half_t*)alloc((size_t)NTOK * 3 * DD * 2);
  float*  SC    = (float*) alloc((size_t)BB * NHH * SS * SS * 4);
  half_t* Ph    = (half_t*)alloc((size_t)BB * NHH * SS * SS * 2);
  half_t* ATTh  = (half_t*)alloc((size_t)NTOK * DD * 2);
  float*  OUTb  = (float*) alloc((size_t)NTOK * DD * 4);
  half_t* Wqkvh = (half_t*)alloc((size_t)NLL * 3 * DD * DD * 2);
  half_t* Woh   = (half_t*)alloc((size_t)NLL * DD * DD * 2);
  float*  MEMO  = (float*) alloc((size_t)NTOK * DDEC * 4);

  convert_w_kernel<<<2048, 256, 0, stream>>>(eqkvw, eow, Wqkvh, Woh);
  embed_kernel<<<NTOK, 256, 0, stream>>>(src, word_emb, syll_emb, X, Xh);

  for (int l = 0; l < NLL; l++){
    // QKV projection: (3072x512) @ (512x1536) -> f16
    gemm_kernel<<<dim3(NTOK/64, (3*DD)/64), 128, 0, stream>>>(
        Xh, DD, Wqkvh + (size_t)l * 3 * DD * DD, DD,
        eqkvb + (size_t)l * 3 * DD, nullptr, QKVh, 3 * DD, DD);
    // attention scores per (b,h)
    score_kernel<<<dim3(SS/64, SS/64, BB*NHH), 128, 0, stream>>>(QKVh, SC);
    softmax_kernel<<<BB*NHH*SS, SS, 0, stream>>>(SC, Ph);
    // P @ V
    pv_kernel<<<dim3(SS/64, HD/64, BB*NHH), 128, 0, stream>>>(Ph, QKVh, ATTh);
    // output projection -> f32
    gemm_kernel<<<dim3(NTOK/64, DD/64), 128, 0, stream>>>(
        ATTh, DD, Woh + (size_t)l * DD * DD, DD,
        eob + (size_t)l * DD, OUTb, nullptr, DD, DD);
    add_ln_kernel<<<NTOK, 256, 0, stream>>>(
        X, OUTb, eln1g + (size_t)l * DD, eln1b + (size_t)l * DD, Xh);
    ffn_ln_kernel<<<NTOK, 256, 0, stream>>>(
        X, eff1w + (size_t)l * FFE * DD, eff1b + (size_t)l * FFE,
        eff2w + (size_t)l * DD * FFE, eff2b + (size_t)l * DD,
        eln2g + (size_t)l * DD, eln2b + (size_t)l * DD, Xh);
  }

  final_mem_kernel<<<NTOK, 256, 0, stream>>>(X, normfg, normfb, eoutw, eoutb,
                                             src, MEMO, out);

  decoder_kernel<<<BB, SS, 0, stream>>>(MEMO,
      dsaqkvw, dsaqkvb, dsaow, dsaob,
      dcaqkvw, dcaqkvb, dcaow, dcaob,
      dln1g, dln1b, dln2g, dln2b, dln3g, dln3b,
      dff1w, dff1b, dff2w, dff2b, out);
}
